// GCN_54511724921371
// MI455X (gfx1250) — matrix operand order repaired and checked by reading, not compiled, not run on hardware
//
#include <hip/hip_runtime.h>
#include <hip/hip_bf16.h>
#include <stdint.h>

// ---------------- types ----------------
typedef __bf16 bf16_t;
typedef bf16_t        v16bf __attribute__((ext_vector_type(16)));
typedef float         v8f   __attribute__((ext_vector_type(8)));
typedef unsigned int  v4u   __attribute__((ext_vector_type(4)));
typedef unsigned int  v8u   __attribute__((ext_vector_type(8)));

#define NN      50000
#define NE      400000
#define KDIM    512
#define K_RANK  409601u   // 1-indexed rank of the median element (Python round-half-even)

// ---------------- helpers ----------------
__device__ __forceinline__ unsigned short f2bf(float f) {
  unsigned u = __float_as_uint(f);
  u += 0x7FFFu + ((u >> 16) & 1u);      // round-to-nearest-even
  return (unsigned short)(u >> 16);
}
__device__ __forceinline__ float bf2f(unsigned short h) {
  return __uint_as_float(((unsigned)h) << 16);
}
__device__ __forceinline__ void atomicAddF32(float* p, float v) {
  __hip_atomic_fetch_add(p, v, __ATOMIC_RELAXED, __HIP_MEMORY_SCOPE_AGENT);
}
__device__ __forceinline__ unsigned f2key(float f) {   // monotonic float->uint
  unsigned b = __float_as_uint(f);
  return (b & 0x80000000u) ? ~b : (b | 0x80000000u);
}
__device__ __forceinline__ v16bf combine16(v4u lo, v4u hi) {
  v8u w;
  w[0]=lo[0]; w[1]=lo[1]; w[2]=lo[2]; w[3]=lo[3];
  w[4]=hi[0]; w[5]=hi[1]; w[6]=hi[2]; w[7]=hi[3];
  return __builtin_bit_cast(v16bf, w);
}

// ---------------- exact median via radix select ----------------
// state: u32[0..255]=hist, [256]=prefix, [257]=remaining rank
__global__ void radix_init(unsigned* st) {
  int t = threadIdx.x;
  if (t < 256) st[t] = 0;
  if (t == 0) { st[256] = 0u; st[257] = K_RANK; }
}
__global__ void radix_hist(const float* __restrict__ S0, const float* __restrict__ S1,
                           const float* __restrict__ S2, const float* __restrict__ S3,
                           int n0, int n1, int n2, int n3,
                           unsigned* st, int shift, unsigned mask) {
  unsigned pref = st[256] & mask;
  int ntot = n0 + n1 + n2 + n3;
  for (int i = blockIdx.x * blockDim.x + threadIdx.x; i < ntot; i += gridDim.x * blockDim.x) {
    float v;
    if (i < n0)           v = S0[i];
    else if (i < n0+n1)   v = S1[i - n0];
    else if (i < n0+n1+n2)v = S2[i - n0 - n1];
    else                  v = S3[i - n0 - n1 - n2];
    unsigned key = f2key(v);
    if ((key & mask) == pref) atomicAdd(&st[(key >> shift) & 0xFFu], 1u);
  }
}
__global__ void radix_scan(unsigned* st, int shift, float* thrOut) {
  // single-thread scan of 256 bins (trivial)
  unsigned r = st[257], cum = 0; int sel = 255;
  for (int b = 0; b < 256; ++b) {
    unsigned c = st[b];
    if (cum + c >= r) { sel = b; break; }
    cum += c;
  }
  st[256] |= ((unsigned)sel) << shift;
  st[257]  = r - cum;
  for (int b = 0; b < 256; ++b) st[b] = 0;
  if (shift == 0) {
    unsigned key  = st[256];
    unsigned bits = (key & 0x80000000u) ? (key & 0x7FFFFFFFu) : ~key;
    *thrOut = __uint_as_float(bits);
  }
}

// ---------------- weight mask + transpose to bf16 ----------------
// Wt[n*K + k] = (S[k,n] > thr) ? bf16(W[k,n]) : 0   (column-major => contiguous-K B fragments)
__global__ void maskw_t(const float* __restrict__ W, const float* __restrict__ S,
                        unsigned short* __restrict__ Wt, int K, int N,
                        const float* __restrict__ thrP) {
  float thr = *thrP;
  int tot = K * N;
  for (int i = blockIdx.x * blockDim.x + threadIdx.x; i < tot; i += gridDim.x * blockDim.x) {
    int k = i / N, n = i % N;
    float w = (S[i] > thr) ? W[i] : 0.0f;
    Wt[(size_t)n * K + k] = f2bf(w);
  }
}

__global__ void cvt_f32_bf16(const float* __restrict__ src, unsigned short* __restrict__ dst, int n) {
  for (int i = blockIdx.x * blockDim.x + threadIdx.x; i < n; i += gridDim.x * blockDim.x)
    dst[i] = f2bf(src[i]);
}
__global__ void zero_u32(unsigned* p, int n) {
  for (int i = blockIdx.x * blockDim.x + threadIdx.x; i < n; i += gridDim.x * blockDim.x) p[i] = 0u;
}
__global__ void zero_f32k(float* p, int n) {
  for (int i = blockIdx.x * blockDim.x + threadIdx.x; i < n; i += gridDim.x * blockDim.x) p[i] = 0.0f;
}

// ---------------- graph normalization ----------------
__global__ void deg_count(const int* __restrict__ ei, unsigned* __restrict__ deg, int E) {
  for (int i = blockIdx.x * blockDim.x + threadIdx.x; i < E; i += gridDim.x * blockDim.x)
    atomicAdd(&deg[ei[E + i]], 1u);
}
__global__ void dinv_k(const unsigned* __restrict__ deg, float* __restrict__ dinv, int n) {
  for (int i = blockIdx.x * blockDim.x + threadIdx.x; i < n; i += gridDim.x * blockDim.x)
    dinv[i] = rsqrtf((float)(deg[i] + 1u));   // +1 self loop; deg>=1 so no clamp needed
}
__global__ void enorm_k(const int* __restrict__ ei, const float* __restrict__ dinv,
                        float* __restrict__ enorm, int E) {
  for (int i = blockIdx.x * blockDim.x + threadIdx.x; i < E; i += gridDim.x * blockDim.x)
    enorm[i] = dinv[ei[i]] * dinv[ei[E + i]];
}

// ---------------- WMMA GEMM: Z[M,N] = A[M,K](bf16) * W (Wt is [N,K] bf16) ----------------
// wave -> 16x64 tile (4 accumulators); 16 K-steps of 32; 4 wmma per step
__global__ void __launch_bounds__(256)
gemm_bf16(const unsigned short* __restrict__ A, const unsigned short* __restrict__ Wt,
          unsigned short* __restrict__ Z, int M, int K, int N) {
  const int lane = threadIdx.x & 31;
  const int wave = blockIdx.x * (blockDim.x >> 5) + (threadIdx.x >> 5);
  const int rowTiles  = M >> 4;
  const int colGroups = N >> 6;
  if (wave >= rowTiles * colGroups) return;      // wave-uniform exit: EXEC stays all-1 for WMMA
  const int rt  = wave % rowTiles;
  const int cg  = wave / rowTiles;
  const int l15 = lane & 15;
  const int kh  = lane >> 4;                      // K-half select: contiguous k = 16*kh + s, same in A and B
  const unsigned short* arow = A  + (size_t)((rt << 4) + l15) * K + (kh << 4);
  const unsigned short* b0p  = Wt + (size_t)(cg * 64 +  0 + l15) * K + (kh << 4);
  const unsigned short* b1p  = b0p + (size_t)16 * K;
  const unsigned short* b2p  = b0p + (size_t)32 * K;
  const unsigned short* b3p  = b0p + (size_t)48 * K;
  v8f acc0 = {}, acc1 = {}, acc2 = {}, acc3 = {};
  for (int k = 0; k < K; k += 32) {
    v16bf a  = combine16(*(const v4u*)(arow + k), *(const v4u*)(arow + k + 8));
    v16bf b0 = combine16(*(const v4u*)(b0p  + k), *(const v4u*)(b0p  + k + 8));
    v16bf b1 = combine16(*(const v4u*)(b1p  + k), *(const v4u*)(b1p  + k + 8));
    v16bf b2 = combine16(*(const v4u*)(b2p  + k), *(const v4u*)(b2p  + k + 8));
    v16bf b3 = combine16(*(const v4u*)(b3p  + k), *(const v4u*)(b3p  + k + 8));
    acc0 = __builtin_amdgcn_wmma_f32_16x16x32_bf16(false, a, false, b0, (short)0, acc0, false, false);
    acc1 = __builtin_amdgcn_wmma_f32_16x16x32_bf16(false, a, false, b1, (short)0, acc1, false, false);
    acc2 = __builtin_amdgcn_wmma_f32_16x16x32_bf16(false, a, false, b2, (short)0, acc2, false, false);
    acc3 = __builtin_amdgcn_wmma_f32_16x16x32_bf16(false, a, false, b3, (short)0, acc3, false, false);
  }
  // C/D layout: VGPR r = row r (lanes 0-15) / row r+8 (lanes 16-31), col = lane%16
  const int colBase = cg * 64 + l15;
  const int rowBase = (rt << 4) + (kh << 3);
  unsigned short* zp = Z + (size_t)rowBase * N + colBase;
#pragma unroll
  for (int r = 0; r < 8; ++r) {
    zp[(size_t)r * N +  0] = f2bf(acc0[r]);
    zp[(size_t)r * N + 16] = f2bf(acc1[r]);
    zp[(size_t)r * N + 32] = f2bf(acc2[r]);
    zp[(size_t)r * N + 48] = f2bf(acc3[r]);
  }
}

// ---------------- aggregation ----------------
// init: out = Z * dinv^2 (self loop) + bias
__global__ void selfloop_bias(const unsigned short* __restrict__ Z, const float* __restrict__ dinv,
                              const float* __restrict__ bias, float* __restrict__ out, int Nn, int D) {
  int tot = Nn * D;
  for (int i = blockIdx.x * blockDim.x + threadIdx.x; i < tot; i += gridDim.x * blockDim.x) {
    int row = i / D, col = i % D;
    float dv = dinv[row];
    out[i] = bf2f(Z[i]) * dv * dv + bias[col];
  }
}
// scatter: out[dst] += Z[src] * enorm  (bf16 gather halves random read traffic, fp32 atomics)
__global__ void edge_scatter(const unsigned short* __restrict__ Z, const int* __restrict__ ei,
                             const float* __restrict__ enorm, float* __restrict__ out,
                             int E, int D) {
  const int chunks = D >> 3;                 // 8 features per thread
  int tot = E * chunks;
  for (int i = blockIdx.x * blockDim.x + threadIdx.x; i < tot; i += gridDim.x * blockDim.x) {
    int e = i / chunks, c = i % chunks;
    int src = ei[e], dst = ei[E + e];
    float nw = enorm[e];
    const unsigned short* zp = Z + (size_t)src * D + (c << 3);
    float* op = out + (size_t)dst * D + (c << 3);
    v4u u = *(const v4u*)zp;                 // 16B coalesced gather
#pragma unroll
    for (int j = 0; j < 4; ++j) {
      unsigned w = u[j];
      atomicAddF32(op + 2*j,     bf2f((unsigned short)(w & 0xFFFFu)) * nw);
      atomicAddF32(op + 2*j + 1, bf2f((unsigned short)(w >> 16))     * nw);
    }
  }
}

// ---------------- PairNorm ----------------
__global__ void stats_col(const float* __restrict__ h, float* __restrict__ colsum,
                          float* __restrict__ colsumsq, int Nn, int D) {
  int j = blockIdx.x * blockDim.x + threadIdx.x;
  if (j >= D) return;
  float s = 0.f, ss = 0.f;
  for (int r = blockIdx.y; r < Nn; r += gridDim.y) {
    float v = h[(size_t)r * D + j];
    s += v; ss += v * v;
  }
  atomicAddF32(&colsum[j],   s);
  atomicAddF32(&colsumsq[j], ss);
}
__global__ void stats_final(const float* __restrict__ colsum, const float* __restrict__ colsumsq,
                            float* __restrict__ colmean, float* __restrict__ sinv, int Nn, int D) {
  __shared__ float red[256];
  float part = 0.f, invN = 1.0f / (float)Nn;
  for (int j = threadIdx.x; j < D; j += blockDim.x) {
    float cs = colsum[j];
    colmean[j] = cs * invN;
    part += colsumsq[j] - cs * cs * invN;    // sum of centered squares
  }
  red[threadIdx.x] = part;
  __syncthreads();
  for (int s = blockDim.x >> 1; s > 0; s >>= 1) {
    if ((int)threadIdx.x < s) red[threadIdx.x] += red[threadIdx.x + s];
    __syncthreads();
  }
  if (threadIdx.x == 0) *sinv = rsqrtf(1e-6f + red[0] * invN);
}
__global__ void pairnorm_relu(const float* __restrict__ h, const float* __restrict__ colmean,
                              const float* __restrict__ sinvP, unsigned short* __restrict__ out,
                              int Nn, int D) {
  float sinv = *sinvP;
  int tot = Nn * D;
  for (int i = blockIdx.x * blockDim.x + threadIdx.x; i < tot; i += gridDim.x * blockDim.x) {
    float v = (h[i] - colmean[i % D]) * sinv;
    out[i] = f2bf(fmaxf(v, 0.0f));
  }
}

// ---------------- host orchestration ----------------
extern "C" void kernel_launch(void* const* d_in, const int* in_sizes, int n_in,
                              void* d_out, int out_size, void* d_ws, size_t ws_size,
                              hipStream_t stream) {
  (void)in_sizes; (void)n_in; (void)out_size; (void)ws_size;
  const float* x  = (const float*)d_in[0];
  const int*   ei = (const int*)d_in[1];
  const float* W[4] = {(const float*)d_in[2], (const float*)d_in[5], (const float*)d_in[8],  (const float*)d_in[11]};
  const float* S[4] = {(const float*)d_in[3], (const float*)d_in[6], (const float*)d_in[9],  (const float*)d_in[12]};
  const float* b[4] = {(const float*)d_in[4], (const float*)d_in[7], (const float*)d_in[10], (const float*)d_in[13]};
  float* out = (float*)d_out;

  const int Nn = NN, E = NE, K = KDIM;
  const int dims[4] = {512, 512, 512, 64};

  // workspace carve (256B aligned)
  char* ws = (char*)d_ws;
  size_t off = 0;
  auto carve = [&](size_t bytes) -> void* {
    void* p = ws + off;
    off = (off + bytes + 255) & ~(size_t)255;
    return p;
  };
  unsigned short* hbf = (unsigned short*)carve((size_t)Nn * 512 * 2);  // activations, bf16
  unsigned short* Zbf = (unsigned short*)carve((size_t)Nn * 512 * 2);  // GEMM output, bf16
  float*          agg = (float*)carve((size_t)Nn * 512 * 4);           // fp32 aggregation
  unsigned short* Wt[4];
  for (int i = 0; i < 4; ++i) Wt[i] = (unsigned short*)carve((size_t)K * dims[i] * 2);
  unsigned* deg   = (unsigned*)carve((size_t)Nn * 4);
  float*    dinv  = (float*)carve((size_t)Nn * 4);
  float*    enorm = (float*)carve((size_t)E * 4);
  float*    colst = (float*)carve(1024 * 4);   // colsum[512] | colsumsq[512]
  float*    colmean = (float*)carve(512 * 4);
  float*    sinv  = (float*)carve(4);
  unsigned* rstate = (unsigned*)carve(258 * 4);
  float*    thrF  = (float*)carve(4);

  // ---- exact median threshold (4-pass radix select over float key bits) ----
  radix_init<<<1, 256, 0, stream>>>(rstate);
  const int n0 = 512*512, n1 = 512*512, n2 = 512*512, n3 = 512*64;
  for (int pass = 0; pass < 4; ++pass) {
    int shift = 24 - 8 * pass;
    unsigned mask = (pass == 0) ? 0u : (0xFFFFFFFFu << (shift + 8));
    radix_hist<<<512, 256, 0, stream>>>(S[0], S[1], S[2], S[3], n0, n1, n2, n3, rstate, shift, mask);
    radix_scan<<<1, 1, 0, stream>>>(rstate, shift, thrF);
  }

  // ---- masked + transposed bf16 weights ----
  for (int i = 0; i < 4; ++i) {
    int tot = K * dims[i];
    maskw_t<<<(tot + 255) / 256, 256, 0, stream>>>(W[i], S[i], Wt[i], K, dims[i], thrF);
  }

  // ---- input activations -> bf16 ----
  cvt_f32_bf16<<<(Nn * 512 + 255) / 256, 256, 0, stream>>>(x, hbf, Nn * 512);

  // ---- degrees / norms ----
  zero_u32<<<(Nn + 255) / 256, 256, 0, stream>>>(deg, Nn);
  deg_count<<<(E + 255) / 256, 256, 0, stream>>>(ei, deg, E);
  dinv_k<<<(Nn + 255) / 256, 256, 0, stream>>>(deg, dinv, Nn);
  enorm_k<<<(E + 255) / 256, 256, 0, stream>>>(ei, dinv, enorm, E);

  // ---- 4 GCN layers ----
  for (int L = 0; L < 4; ++L) {
    int D = dims[L];
    int waves  = (Nn / 16) * (D / 64);
    int blocks = (waves + 7) / 8;                       // 8 waves (256 threads) per block
    gemm_bf16<<<blocks, 256, 0, stream>>>(hbf, Wt[L], Zbf, Nn, K, D);

    float* tgt = (L == 3) ? out : agg;
    selfloop_bias<<<(Nn * D + 255) / 256, 256, 0, stream>>>(Zbf, dinv, b[L], tgt, Nn, D);
    edge_scatter<<<(E * (D / 8) + 255) / 256, 256, 0, stream>>>(Zbf, ei, enorm, tgt, E, D);

    if (L < 3) {
      zero_f32k<<<4, 256, 0, stream>>>(colst, 1024);
      stats_col<<<dim3(D / 256, 64), 256, 0, stream>>>(agg, colst, colst + 512, Nn, D);
      stats_final<<<1, 256, 0, stream>>>(colst, colst + 512, colmean, sinv, Nn, D);
      pairnorm_relu<<<(Nn * D + 255) / 256, 256, 0, stream>>>(agg, colmean, sinv, hbf, Nn, D);
    }
  }
}
